// FeedForwardQuantum_65481071396242
// MI455X (gfx1250) — matrix operand order repaired and checked by reading, not compiled, not run on hardware
//
#include <hip/hip_runtime.h>
#include <hip/hip_bf16.h>

typedef __attribute__((ext_vector_type(16))) _Float16 v16h;
typedef __attribute__((ext_vector_type(8)))  _Float16 v8h;
typedef __attribute__((ext_vector_type(4)))  _Float16 v4h;
typedef __attribute__((ext_vector_type(8)))  float    v8f;
typedef __attribute__((ext_vector_type(4)))  unsigned int u32x4;
typedef __attribute__((ext_vector_type(8)))  int          i32x8;
typedef __attribute__((ext_vector_type(4)))  int          i32x4;

union ABFrag { v16h v; v8h h[2]; };

#define EMBED 1024
#define FFN   4096
#define NQ    8
#define BM    128   // tokens per workgroup
#define BN    256   // embed cols per workgroup
#define BK    64    // ffn (K) per staged tile (2 WMMA sub-steps of 32)
#define HSTR  72    // f16 row stride = 144B = 36 dwords -> conflict-free 16-row frag reads

// ---------------------------------------------------------------------------
// Prep: W2 fp32 -> fp16 (done once; W2 is reused by all 1024 GEMM workgroups)
// ---------------------------------------------------------------------------
__global__ __launch_bounds__(256)
void w2_to_f16_kernel(const float* __restrict__ W2, _Float16* __restrict__ w2h) {
    const int n4 = (FFN * EMBED) / 4;
    const int stride = gridDim.x * 256;
    for (int i = blockIdx.x * 256 + threadIdx.x; i < n4; i += stride) {
        const float4 w = ((const float4*)W2)[i];
        v4h o;
        o[0] = (_Float16)w.x;  o[1] = (_Float16)w.y;
        o[2] = (_Float16)w.z;  o[3] = (_Float16)w.w;
        ((v4h*)w2h)[i] = o;
    }
}

// ---------------------------------------------------------------------------
// TDM: issue tensor_load_to_lds of a 64(k) x 256(n) fp16 tile of W2h.
// D# per CDNA5 ISA:
//   group0: count=1 | lds_addr | global_addr(57b) | type=2
//   group1: data_size=1(2B) | pad_enable, pad_interval=4(32 dw), pad_amount=3(4 dw)
//           tensor_dim0=4096, tensor_dim1=1024, tile_dim0=64, tile_dim1=256,
//           tensor_dim0_stride=4096  -> rows land at 144B LDS stride (HSTR)
// ---------------------------------------------------------------------------
__device__ __forceinline__ void tdm_load_w2_tile(const _Float16* gsrc, void* ldst) {
    const unsigned long long ga = (unsigned long long)(uintptr_t)gsrc;
    const unsigned lds_addr = (unsigned)(uintptr_t)ldst;   // flat addr low 32b == LDS offset
    u32x4 g0 = { 1u,                                        // count=1 (valid descriptor)
                 lds_addr,
                 (unsigned)ga,
                 ((unsigned)(ga >> 32) & 0x01FFFFFFu) | (2u << 30) };  // type=2 ("image")
    i32x8 g1 = { (int)((1u << 16) |        // data_size = 2 bytes
                       (1u << 20) |        // pad_enable
                       (4u << 22) |        // pad_interval: 32 dwords (=128B = 64 f16 row)
                       (3u << 25)),        // pad_amount: 4 dwords (=16B -> 144B stride)
                 (int)(4096u << 16),       // tensor_dim0 = 4096  (bits[79:48] low part)
                 (int)(1024u << 16),       // tensor_dim0 hi=0 | tensor_dim1 = 1024
                 (int)(64u << 16),         // tensor_dim1 hi=0 | tile_dim0 = 64
                 (int)(256u),              // tile_dim1 = 256 | tile_dim2 = 0 (2-D)
                 (int)(4096u),             // tensor_dim0_stride = 4096 elements
                 0, 0 };                   // stride0 hi | tensor_dim1_stride = 0 (unused, 2-D)
    i32x4 gz4 = { 0, 0, 0, 0 };            // groups 2/3 unused for 2-D tile
    i32x8 gz8 = { 0, 0, 0, 0, 0, 0, 0, 0 };
    __builtin_amdgcn_tensor_load_to_lds(g0, g1, gz4, gz4, gz8, 0);
}

// ---------------------------------------------------------------------------
// Main fused kernel, W2 tiles staged by the Tensor Data Mover
// ---------------------------------------------------------------------------
__global__ __launch_bounds__(256)
void ffq_tdm_kernel(const float* __restrict__ x,
                    const float* __restrict__ theta,
                    const float* __restrict__ W1,
                    const float* __restrict__ b1,
                    const _Float16* __restrict__ w2h,
                    const float* __restrict__ b2,
                    float* __restrict__ out)
{
    __shared__ float lds_q[BM][NQ];                    // 4 KB
    __shared__ float lds_w1[BK][NQ];                   // 2 KB
    __shared__ float lds_b1[BK];                       // 256 B
    __shared__ __align__(16) _Float16 lds_h[BM][HSTR]; // 18 KB  A tile (h, fp16)
    __shared__ __align__(16) _Float16 lds_b[BN][HSTR]; // 36 KB  B tile (W2h via TDM)

    const int tid  = threadIdx.x;
    const int lane = tid & 31;
    const int lr   = lane & 15;
    const int g    = lane >> 4;
    const int wave = tid >> 5;
    const int wm   = wave >> 2;       // 0..1 : 64-row group
    const int wn   = wave & 3;        // 0..3 : 64-col group

    const int t0    = blockIdx.x * BM;
    const int nbase = blockIdx.y * BN;

    // ---- q tile: 128 tokens x 8 wires ----
    {
        const int tok = tid >> 1;
        const int ig  = (tid & 1) * 4;
        const float4 xv = *(const float4*)(x + (size_t)(t0 + tok) * EMBED + ig);
        lds_q[tok][ig + 0] = cosf(theta[ig + 0]) * cosf(xv.x);
        lds_q[tok][ig + 1] = cosf(theta[ig + 1]) * cosf(xv.y);
        lds_q[tok][ig + 2] = cosf(theta[ig + 2]) * cosf(xv.z);
        lds_q[tok][ig + 3] = cosf(theta[ig + 3]) * cosf(xv.w);
    }
    __syncthreads();

    const int hm   = tid >> 1;
    const int half = tid & 1;
    float qv[NQ];
#pragma unroll
    for (int i = 0; i < NQ; ++i) qv[i] = lds_q[hm][i];

    v8f acc[4][4];
#pragma unroll
    for (int am = 0; am < 4; ++am)
#pragma unroll
        for (int nt = 0; nt < 4; ++nt)
#pragma unroll
            for (int v = 0; v < 8; ++v) acc[am][nt][v] = 0.0f;

    for (int k0 = 0; k0 < FFN; k0 += BK) {
        __syncthreads();   // all waves done reading lds_b / lds_h of previous tile

        // ---- wave 0: kick the TDM for this K tile (overlaps with h compute) ----
        if (wave == 0)
            tdm_load_w2_tile(w2h + (size_t)nbase * FFN + k0, &lds_b[0][0]);

        // ---- stage W1 k-tile (64x8 fp32) + b1 ----
        ((float*)lds_w1)[tid]       = W1[(size_t)k0 * NQ + tid];
        ((float*)lds_w1)[tid + 256] = W1[(size_t)k0 * NQ + 256 + tid];
        if (tid < BK) lds_b1[tid] = b1[k0 + tid];
        __syncthreads();

        // ---- compute h tile (128 x 64) = relu(q.W1^T + b1), fp16 into LDS ----
#pragma unroll
        for (int jj = 0; jj < 32; jj += 2) {
            const int j = half * 32 + jj;
            float s0 = lds_b1[j];
            float s1 = lds_b1[j + 1];
#pragma unroll
            for (int i = 0; i < NQ; ++i) {
                s0 = fmaf(qv[i], lds_w1[j][i],     s0);
                s1 = fmaf(qv[i], lds_w1[j + 1][i], s1);
            }
            s0 = fmaxf(s0, 0.0f);
            s1 = fmaxf(s1, 0.0f);
            lds_h[hm][j]     = (_Float16)s0;
            lds_h[hm][j + 1] = (_Float16)s1;
        }

        if (wave == 0)
            __builtin_amdgcn_s_wait_tensorcnt(0);   // W2h tile landed in LDS
        __syncthreads();                            // h + W2 tiles visible to all waves

        // ---- two K=32 WMMA sub-steps over the staged 64-wide tile ----
#pragma unroll
        for (int s = 0; s < 2; ++s) {
            const int kb = 32 * s;
            ABFrag afrag[4];
#pragma unroll
            for (int am = 0; am < 4; ++am) {
                const int row = wm * 64 + am * 16 + lr;
                afrag[am].h[0] = *(const v8h*)&lds_h[row][kb + 8 * g];
                afrag[am].h[1] = *(const v8h*)&lds_h[row][kb + 16 + 8 * g];
            }
            ABFrag bfrag[4];
#pragma unroll
            for (int nt = 0; nt < 4; ++nt) {
                const int col = wn * 64 + nt * 16 + lr;
                bfrag[nt].h[0] = *(const v8h*)&lds_b[col][kb + 8 * g];
                bfrag[nt].h[1] = *(const v8h*)&lds_b[col][kb + 16 + 8 * g];
            }
#pragma unroll
            for (int am = 0; am < 4; ++am)
#pragma unroll
                for (int nt = 0; nt < 4; ++nt)
                    acc[am][nt] = __builtin_amdgcn_wmma_f32_16x16x32_f16(
                        false, afrag[am].v, false, bfrag[nt].v,
                        (short)0, acc[am][nt], false, false);
        }
    }

    // ---- epilogue: +b2, store fp32 ----
#pragma unroll
    for (int nt = 0; nt < 4; ++nt) {
        const int ncol = nbase + wn * 64 + nt * 16 + lr;
        const float b2v = b2[ncol];
#pragma unroll
        for (int am = 0; am < 4; ++am) {
            const int rbase = t0 + wm * 64 + am * 16 + 8 * g;
#pragma unroll
            for (int v = 0; v < 8; ++v)
                out[(size_t)(rbase + v) * EMBED + ncol] = acc[am][nt][v] + b2v;
        }
    }
}

// ---------------------------------------------------------------------------
// Fallback (no workspace): round-2 kernel with VALU cvt staging of W2
// ---------------------------------------------------------------------------
__global__ __launch_bounds__(256)
void ffq_fused_cvt_kernel(const float* __restrict__ x,
                          const float* __restrict__ theta,
                          const float* __restrict__ W1,
                          const float* __restrict__ b1,
                          const float* __restrict__ W2,
                          const float* __restrict__ b2,
                          float* __restrict__ out)
{
    __shared__ float lds_q[BM][NQ];
    __shared__ float lds_w1[BK][NQ];
    __shared__ float lds_b1[BK];
    __shared__ __align__(16) _Float16 lds_h[BM][HSTR];
    __shared__ __align__(16) _Float16 lds_b[BN][HSTR];

    const int tid  = threadIdx.x;
    const int lane = tid & 31;
    const int lr   = lane & 15;
    const int g    = lane >> 4;
    const int wave = tid >> 5;
    const int wm   = wave >> 2;
    const int wn   = wave & 3;

    const int t0    = blockIdx.x * BM;
    const int nbase = blockIdx.y * BN;

    {
        const int tok = tid >> 1;
        const int ig  = (tid & 1) * 4;
        const float4 xv = *(const float4*)(x + (size_t)(t0 + tok) * EMBED + ig);
        lds_q[tok][ig + 0] = cosf(theta[ig + 0]) * cosf(xv.x);
        lds_q[tok][ig + 1] = cosf(theta[ig + 1]) * cosf(xv.y);
        lds_q[tok][ig + 2] = cosf(theta[ig + 2]) * cosf(xv.z);
        lds_q[tok][ig + 3] = cosf(theta[ig + 3]) * cosf(xv.w);
    }
    __syncthreads();

    const int hm   = tid >> 1;
    const int half = tid & 1;
    float qv[NQ];
#pragma unroll
    for (int i = 0; i < NQ; ++i) qv[i] = lds_q[hm][i];

    v8f acc[4][4];
#pragma unroll
    for (int am = 0; am < 4; ++am)
#pragma unroll
        for (int nt = 0; nt < 4; ++nt)
#pragma unroll
            for (int v = 0; v < 8; ++v) acc[am][nt][v] = 0.0f;

    const int w2_n  = tid >> 4;
    const int w2_kq = tid & 15;

    for (int k0 = 0; k0 < FFN; k0 += BK) {
        __syncthreads();

        ((float*)lds_w1)[tid]       = W1[(size_t)k0 * NQ + tid];
        ((float*)lds_w1)[tid + 256] = W1[(size_t)k0 * NQ + 256 + tid];
        if (tid < BK) lds_b1[tid] = b1[k0 + tid];

#pragma unroll
        for (int r = 0; r < 16; ++r) {
            const int n = w2_n + 16 * r;
            const float* src = W2 + (size_t)(nbase + n) * FFN + k0 + w2_kq * 4;
            const float4 w = *(const float4*)src;
            _Float16* dst = &lds_b[n][w2_kq * 4];
            dst[0] = (_Float16)w.x;  dst[1] = (_Float16)w.y;
            dst[2] = (_Float16)w.z;  dst[3] = (_Float16)w.w;
            if (k0 + BK < FFN)
                __builtin_prefetch(src + BK, 0, 1);
        }
        __syncthreads();

#pragma unroll
        for (int jj = 0; jj < 32; jj += 2) {
            const int j = half * 32 + jj;
            float s0 = lds_b1[j];
            float s1 = lds_b1[j + 1];
#pragma unroll
            for (int i = 0; i < NQ; ++i) {
                s0 = fmaf(qv[i], lds_w1[j][i],     s0);
                s1 = fmaf(qv[i], lds_w1[j + 1][i], s1);
            }
            s0 = fmaxf(s0, 0.0f);
            s1 = fmaxf(s1, 0.0f);
            lds_h[hm][j]     = (_Float16)s0;
            lds_h[hm][j + 1] = (_Float16)s1;
        }
        __syncthreads();

#pragma unroll
        for (int s = 0; s < 2; ++s) {
            const int kb = 32 * s;
            ABFrag afrag[4];
#pragma unroll
            for (int am = 0; am < 4; ++am) {
                const int row = wm * 64 + am * 16 + lr;
                afrag[am].h[0] = *(const v8h*)&lds_h[row][kb + 8 * g];
                afrag[am].h[1] = *(const v8h*)&lds_h[row][kb + 16 + 8 * g];
            }
            ABFrag bfrag[4];
#pragma unroll
            for (int nt = 0; nt < 4; ++nt) {
                const int col = wn * 64 + nt * 16 + lr;
                bfrag[nt].h[0] = *(const v8h*)&lds_b[col][kb + 8 * g];
                bfrag[nt].h[1] = *(const v8h*)&lds_b[col][kb + 16 + 8 * g];
            }
#pragma unroll
            for (int am = 0; am < 4; ++am)
#pragma unroll
                for (int nt = 0; nt < 4; ++nt)
                    acc[am][nt] = __builtin_amdgcn_wmma_f32_16x16x32_f16(
                        false, afrag[am].v, false, bfrag[nt].v,
                        (short)0, acc[am][nt], false, false);
        }
    }

#pragma unroll
    for (int nt = 0; nt < 4; ++nt) {
        const int ncol = nbase + wn * 64 + nt * 16 + lr;
        const float b2v = b2[ncol];
#pragma unroll
        for (int am = 0; am < 4; ++am) {
            const int rbase = t0 + wm * 64 + am * 16 + 8 * g;
#pragma unroll
            for (int v = 0; v < 8; ++v)
                out[(size_t)(rbase + v) * EMBED + ncol] = acc[am][nt][v] + b2v;
        }
    }
}

extern "C" void kernel_launch(void* const* d_in, const int* in_sizes, int n_in,
                              void* d_out, int out_size, void* d_ws, size_t ws_size,
                              hipStream_t stream) {
    const float* x     = (const float*)d_in[0];
    const float* theta = (const float*)d_in[1];
    const float* W1    = (const float*)d_in[2];
    const float* b1    = (const float*)d_in[3];
    const float* W2    = (const float*)d_in[4];
    const float* b2    = (const float*)d_in[5];
    float* out = (float*)d_out;

    const int tokens = in_sizes[0] / EMBED;          // 32768
    dim3 grid(tokens / BM, EMBED / BN);              // (256, 4)
    dim3 block(256);

    const size_t w2h_bytes = (size_t)FFN * EMBED * sizeof(_Float16);  // 8 MB
    if (ws_size >= w2h_bytes) {
        _Float16* w2h = (_Float16*)d_ws;
        w2_to_f16_kernel<<<1024, 256, 0, stream>>>(W2, w2h);
        ffq_tdm_kernel<<<grid, block, 0, stream>>>(x, theta, W1, b1, w2h, b2, out);
    } else {
        ffq_fused_cvt_kernel<<<grid, block, 0, stream>>>(x, theta, W1, b1, W2, b2, out);
    }
}
